// GNNEncoder_24592982737095
// MI455X (gfx1250) — matrix operand order repaired and checked by reading, not compiled, not run on hardware
//
#include <hip/hip_runtime.h>
#include <hip/hip_bf16.h>
#include <math.h>

// HGT encoder for MI455X (gfx1250): LDS-staged bf16 WMMA GEMMs + atomic scatter softmax.

#define Hh   8
#define Dh   32
#define HID  256
#define NNODES 50000
#define NEDGES 500000
#define BN   64                 // columns per GEMM block
#define LDKMAX 272              // 256 + 16 pad (keeps 32B alignment, spreads banks)

typedef __attribute__((ext_vector_type(16))) __bf16 v16bf;
typedef __attribute__((ext_vector_type(8)))  float  v8f;

__device__ __forceinline__ __bf16 tobf(float f) { return (__bf16)f; }

// ordered-uint encoding so unsigned atomicMax implements float max
__device__ __forceinline__ unsigned f32ord(float f) {
  unsigned u = __float_as_uint(f);
  return (u & 0x80000000u) ? ~u : (u | 0x80000000u);
}
__device__ __forceinline__ float ordf32(unsigned u) {
  u = (u & 0x80000000u) ? (u & 0x7FFFFFFFu) : ~u;
  return __uint_as_float(u);
}

// Build 16-bit A fragment: lanes 0-15 row=lane K={0..15},
// lanes 16-31 K={16..31}; one contiguous 64B run -> 4x float4 loads.
__device__ __forceinline__ v16bf load_a_frag(const float* __restrict__ base) {
  const float4* p = (const float4*)base;   // base already at (row, kk + kbase)
  float4 a0 = p[0], a1 = p[1], a2 = p[2], a3 = p[3];
  v16bf a;
  a[0]=tobf(a0.x); a[1]=tobf(a0.y); a[2]=tobf(a0.z); a[3]=tobf(a0.w);
  a[4]=tobf(a1.x); a[5]=tobf(a1.y); a[6]=tobf(a1.z); a[7]=tobf(a1.w);
  a[8]=tobf(a2.x); a[9]=tobf(a2.y); a[10]=tobf(a2.z); a[11]=tobf(a2.w);
  a[12]=tobf(a3.x); a[13]=tobf(a3.y); a[14]=tobf(a3.z); a[15]=tobf(a3.w);
  return a;
}

// ---------------- C[M,256] = act(A[M,K] @ W[K,256] + bias) ------------------
// Block = 8 waves. Block stages W[:, col0:col0+64] in LDS (bf16, column-major
// so each lane's B fragment is one contiguous 32B LDS read). Each wave owns a
// 16-row tile and computes 16x64 (4 WMMAs per K-step sharing one A fragment).
__global__ __launch_bounds__(256) void gemm_bias_kernel(
    const float* __restrict__ A, const float* __restrict__ W,
    const float* __restrict__ bias, float* __restrict__ C,
    int M, int K, int act)
{
  __shared__ __bf16 lw[BN * LDKMAX];       // 34 KB
  const int ldk = K + 16;                  // 32B-aligned column stride
  const int tid  = threadIdx.x;
  const int wave = tid >> 5;
  const int lane = tid & 31;
  const int nblk = blockIdx.x & 3;         // 64-col slice
  const int mblk = blockIdx.x >> 2;        // 128-row slice
  const int col0 = nblk * BN;

  // cooperative, coalesced load of W slice; swizzle to [col][k] bf16
  for (int idx = tid; idx < BN * K; idx += 256) {
    int k = idx >> 6, c = idx & 63;
    lw[c * ldk + k] = tobf(W[(long)k * HID + col0 + c]);
  }
  __syncthreads();

  const int hi  = (lane >= 16);
  const int l15 = lane & 15;
  const int tm  = mblk * 8 + wave;
  if (tm * 16 >= M) return;
  const int arow  = tm * 16 + l15;
  const int kbase = hi ? 16 : 0;
  const int kbl   = hi ? 16 : 0;

  v8f acc[4] = {};
  for (int kk = 0; kk < K; kk += 32) {
    v16bf a = load_a_frag(A + (long)arow * K + kk + kbase);
#pragma unroll
    for (int nt = 0; nt < 4; ++nt) {
      const v16bf b = *(const v16bf*)(lw + (nt * 16 + l15) * ldk + kk + kbl);
      acc[nt] = __builtin_amdgcn_wmma_f32_16x16x32_bf16(false, a, false, b,
                                                        (short)0, acc[nt], false, false);
    }
  }
#pragma unroll
  for (int nt = 0; nt < 4; ++nt) {
    const int bcol = col0 + nt * 16 + l15;
    const float bv = bias[bcol];
#pragma unroll
    for (int i = 0; i < 8; ++i) {
      int ro = tm * 16 + (hi ? (i + 8) : i);
      float v = acc[nt][i] + bv;
      if (act == 1) v = fmaxf(v, 0.0f);
      C[(long)ro * HID + bcol] = v;
    }
  }
}

// ------- Y[n,h,:] = X[n,h,:] @ R[h]  (per-head 32x32 relation transform) ----
// Block stages all 8 heads' 32x32 matrices in LDS (16 KB bf16, column-major).
// Wave w handles head w of one 16-row tile: one A fragment, two WMMAs.
__global__ __launch_bounds__(256) void rel_kernel(
    const float* __restrict__ X, const float* __restrict__ R,
    float* __restrict__ Y, int M)
{
  __shared__ __bf16 lr[Hh * Dh * Dh];      // [h][col][k], 16 KB
  const int tid  = threadIdx.x;
  const int wave = tid >> 5;
  const int lane = tid & 31;

  for (int idx = tid; idx < Hh * Dh * Dh; idx += 256) {
    int h = idx >> 10, rem = idx & 1023;
    int k = rem >> 5, c = rem & 31;
    lr[(h * Dh + c) * Dh + k] = tobf(R[h * Dh * Dh + k * Dh + c]);
  }
  __syncthreads();

  const int h  = wave;                     // 8 waves = 8 heads
  const int tm = blockIdx.x;
  if (tm * 16 >= M) return;

  const int hi  = (lane >= 16);
  const int l15 = lane & 15;
  const int kbase = hi ? 16 : 0;
  const int kbl   = hi ? 16 : 0;

  v16bf a = load_a_frag(X + (long)(tm * 16 + l15) * HID + h * Dh + kbase);
#pragma unroll
  for (int nt = 0; nt < 2; ++nt) {
    const v16bf b = *(const v16bf*)(lr + (h * Dh + nt * 16 + l15) * Dh + kbl);
    v8f acc = {};
    acc = __builtin_amdgcn_wmma_f32_16x16x32_bf16(false, a, false, b,
                                                  (short)0, acc, false, false);
#pragma unroll
    for (int i = 0; i < 8; ++i) {
      int ro = tm * 16 + (hi ? (i + 8) : i);
      Y[(long)ro * HID + h * Dh + nt * 16 + l15] = acc[i];
    }
  }
}

// ------------------------------ elementwise ---------------------------------
__global__ void fill_f32(float* __restrict__ p, float v, long n) {
  long i = (long)blockIdx.x * blockDim.x + threadIdx.x;
  if (i < n) p[i] = v;
}
__global__ void fill_u32(unsigned* __restrict__ p, unsigned v, long n) {
  long i = (long)blockIdx.x * blockDim.x + threadIdx.x;
  if (i < n) p[i] = v;
}
__global__ void gelu_kernel(const float* __restrict__ x, float* __restrict__ y, long n) {
  long i = (long)blockIdx.x * blockDim.x + threadIdx.x;
  if (i < n) {
    float v = x[i];
    y[i] = 0.5f * v * (1.0f + erff(v * 0.70710678118654752f));
  }
}
__global__ void blend_kernel(const float* __restrict__ o, const float* __restrict__ x,
                             const float* __restrict__ skip, float* __restrict__ out, long n) {
  long i = (long)blockIdx.x * blockDim.x + threadIdx.x;
  if (i < n) {
    float beta = 1.0f / (1.0f + expf(-skip[0]));
    out[i] = beta * o[i] + (1.0f - beta) * x[i];
  }
}

// --------------------------- edge-phase kernels -----------------------------
// thread t -> (edge e = t/8, head h = t%8). ei = [2, E] int32 (src row, dst row)
__global__ __launch_bounds__(256) void alpha_kernel(
    const int* __restrict__ ei, int E,
    const float* __restrict__ Q, const float* __restrict__ KREL,
    const float* __restrict__ prel,
    float* __restrict__ alpha, unsigned* __restrict__ amax)
{
  int t = blockIdx.x * 256 + threadIdx.x;
  if (t >= E * Hh) return;
  int e = t >> 3, h = t & 7;
  int src = ei[e], dst = ei[E + e];
  const float4* qr = (const float4*)(Q + (long)dst * HID + h * Dh);
  const float4* kr = (const float4*)(KREL + (long)src * HID + h * Dh);
  __builtin_prefetch(kr, 0, 1);
  float s = 0.f;
#pragma unroll
  for (int i = 0; i < 8; ++i) {
    float4 qa = qr[i], ka = kr[i];
    s += qa.x * ka.x + qa.y * ka.y + qa.z * ka.z + qa.w * ka.w;
  }
  s *= prel[h] * 0.17677669529663688f;   // prel / sqrt(32)
  alpha[t] = s;
  atomicMax(&amax[(long)dst * Hh + h], f32ord(s));
}

__global__ __launch_bounds__(256) void expsum_kernel(
    const int* __restrict__ ei, int E, float* __restrict__ alpha,
    const unsigned* __restrict__ amax, float* __restrict__ denom)
{
  int t = blockIdx.x * 256 + threadIdx.x;
  if (t >= E * Hh) return;
  int e = t >> 3, h = t & 7;
  int dst = ei[E + e];
  float m = ordf32(amax[(long)dst * Hh + h]);
  float ex = expf(alpha[t] - m);
  alpha[t] = ex;
  atomicAdd(&denom[(long)dst * Hh + h], ex);
}

__global__ __launch_bounds__(256) void scatter_kernel(
    const int* __restrict__ ei, int E, const float* __restrict__ alpha,
    const float* __restrict__ denom, const float* __restrict__ VREL,
    float* __restrict__ ACC)
{
  int t = blockIdx.x * 256 + threadIdx.x;
  if (t >= E * Hh) return;
  int e = t >> 3, h = t & 7;
  int src = ei[e], dst = ei[E + e];
  float a = alpha[t] / (denom[(long)dst * Hh + h] + 1e-16f);
  const float* vr = VREL + (long)src * HID + h * Dh;
  __builtin_prefetch(vr, 0, 1);
  float* ar = ACC + (long)dst * HID + h * Dh;
#pragma unroll
  for (int d = 0; d < Dh; ++d) atomicAdd(&ar[d], vr[d] * a);
}

// ------------------------------- host side ----------------------------------
extern "C" void kernel_launch(void* const* d_in, const int* in_sizes, int n_in,
                              void* d_out, int out_size, void* d_ws, size_t ws_size,
                              hipStream_t stream) {
  (void)in_sizes; (void)n_in; (void)out_size; (void)ws_size;
  const float* x_author = (const float*)d_in[0];
  const float* x_paper  = (const float*)d_in[1];
  const int* ei[3] = { (const int*)d_in[2], (const int*)d_in[3], (const int*)d_in[4] };

  const long NB = (long)NNODES * HID;           // 12.8M floats per node tensor
  float* ws = (float*)d_ws;
  float* X[2]   = { ws + 0 * NB, ws + 1 * NB };
  float* Kb[2]  = { ws + 2 * NB, ws + 3 * NB };
  float* Qb[2]  = { ws + 4 * NB, ws + 5 * NB };
  float* Vb[2]  = { ws + 6 * NB, ws + 7 * NB };
  float* KREL   = ws + 8 * NB;
  float* VREL   = ws + 9 * NB;
  float* ACC[2] = { ws + 10 * NB, ws + 11 * NB };
  float* ALPHA  = ws + 12 * NB;                 // E*H = 4M floats
  float* DEN    = ALPHA + (long)NEDGES * Hh;    // N*H
  unsigned* AMAX = (unsigned*)(DEN + (long)NNODES * Hh);

  dim3 blk(256);
  const int mt = (NNODES + 15) / 16;            // 3125 row tiles
  const int mblocks = (mt + 7) / 8;             // 8 row-tiles per block
  const int gemm_blocks = mblocks * 4;          // x4 column slices of 64
  const int rel_blocks  = mt;                   // one block per 16-row tile (8 heads)
  const int EH_blocks   = (NEDGES * Hh + 255) / 256;
  const long NH = (long)NNODES * Hh;
  const int NH_blocks = (int)((NH + 255) / 256);
  const int NB_blocks = (int)((NB + 255) / 256);

  auto gemm = [&](const float* A, int widx, float* C, int K, int act) {
    gemm_bias_kernel<<<gemm_blocks, blk, 0, stream>>>(
        A, (const float*)d_in[widx], (const float*)d_in[widx + 1], C, NNODES, K, act);
  };

  // input projections + ReLU
  gemm(x_author, 5, X[0], 128, 1);
  gemm(x_paper,  7, X[1], 256, 1);

  const int st[3] = { 0, 1, 1 };   // writes: author->paper, written_by: paper->author, cites: paper->paper
  const int dt[3] = { 1, 0, 1 };

  for (int l = 0; l < 2; ++l) {
    int base = 9 + 27 * l;
    for (int t = 0; t < 2; ++t) {
      int pb = base + 9 * t;   // k_w,k_b,q_w,q_b,v_w,v_b,a_w,a_b,skip
      gemm(X[t], pb + 0, Kb[t], 256, 0);
      gemm(X[t], pb + 2, Qb[t], 256, 0);
      gemm(X[t], pb + 4, Vb[t], 256, 0);
      fill_f32<<<NB_blocks, blk, 0, stream>>>(ACC[t], 0.0f, NB);
    }
    for (int et = 0; et < 3; ++et) {
      int eb = base + 18 + 3 * et;  // arel, mrel, prel
      rel_kernel<<<rel_blocks, blk, 0, stream>>>(Kb[st[et]], (const float*)d_in[eb],     KREL, NNODES);
      rel_kernel<<<rel_blocks, blk, 0, stream>>>(Vb[st[et]], (const float*)d_in[eb + 1], VREL, NNODES);
      fill_u32<<<NH_blocks, blk, 0, stream>>>(AMAX, 0x007FFFFFu, NH);  // ord(-inf)
      fill_f32<<<NH_blocks, blk, 0, stream>>>(DEN, 0.0f, NH);
      alpha_kernel<<<EH_blocks, blk, 0, stream>>>(ei[et], NEDGES, Qb[dt[et]], KREL,
                                                  (const float*)d_in[eb + 2], ALPHA, AMAX);
      expsum_kernel<<<EH_blocks, blk, 0, stream>>>(ei[et], NEDGES, ALPHA, AMAX, DEN);
      scatter_kernel<<<EH_blocks, blk, 0, stream>>>(ei[et], NEDGES, ALPHA, DEN, VREL, ACC[dt[et]]);
    }
    for (int t = 0; t < 2; ++t) {
      int pb = base + 9 * t;
      gelu_kernel<<<NB_blocks, blk, 0, stream>>>(ACC[t], KREL, NB);   // KREL free now
      gemm(KREL, pb + 6, VREL, 256, 0);                               // VREL = o
      float* dest = (l == 1) ? ((float*)d_out + (long)t * NB) : X[t];
      blend_kernel<<<NB_blocks, blk, 0, stream>>>(VREL, X[t], (const float*)d_in[pb + 8], dest, NB);
    }
  }
}
